// ZImageTransformer2DModel_69123203662441
// MI455X (gfx1250) — matrix operand order repaired and checked by reading, not compiled, not run on hardware
//
#include <hip/hip_runtime.h>
#include <hip/hip_bf16.h>
#include <math.h>

// ---------------- problem constants ----------------
#define Bc     2
#define Sc     2048
#define Dc     2048
#define Hc     16
#define HDc    128
#define FFc    5461
#define FFp    5504            // padded FF: 43*128, multiple of 32
#define ADALNc 256
#define Mrows  (Bc*Sc)         // 4096
#define EPSc   1e-5f

typedef __bf16 bf16_t;
typedef __attribute__((ext_vector_type(16))) __bf16 v16bf;
typedef __attribute__((ext_vector_type(8)))  float  v8f;

__device__ __forceinline__ v8f v8f_zero() {
    v8f z = {0.f,0.f,0.f,0.f,0.f,0.f,0.f,0.f};
    return z;
}

// ---------------------------------------------------------------------------
// CDNA5 async global->LDS copy (GLOBAL_LOAD_ASYNC_TO_LDS_B128, ASYNCcnt).
// Per lane: copies 16 bytes from the lane's global address to the lane's LDS
// byte address (VDST VGPR). Generic LDS pointers carry the LDS offset in the
// low 32 bits (aperture base lives in the high half), so a truncating cast
// yields the DS address the instruction wants.
// ---------------------------------------------------------------------------
__device__ __forceinline__ void async_copy_b128(bf16_t* lds_dst, const bf16_t* gsrc) {
    unsigned int loff = (unsigned int)(size_t)lds_dst;
    asm volatile("global_load_async_to_lds_b128 %0, %1, off"
                 :: "v"(loff), "v"(gsrc)
                 : "memory");
}

__device__ __forceinline__ void wait_async() {
#if defined(__has_builtin)
#if __has_builtin(__builtin_amdgcn_s_wait_asynccnt)
    __builtin_amdgcn_s_wait_asynccnt(0);
#else
    asm volatile("s_wait_asynccnt 0x0" ::: "memory");
#endif
#else
    asm volatile("s_wait_asynccnt 0x0" ::: "memory");
#endif
}

// ---------------------------------------------------------------------------
// WMMA fragment loaders (bf16, 16x16x32, wave32).
// A fragment (16x32, M x K): lanes 0-15 hold row M=lane, K in {0..7, 16..23};
//                            lanes 16-31 hold row M=lane-16, K in {8..15, 24..31}.
// B fragment (32x16, K x N) stored as Bt[n][k] (K-major): lanes 0-15 hold
//   col N=lane, K in {0..7, 16..23}; lanes 16-31 hold col N=lane-16, K in {8..15, 24..31}.
// Both load as two 16-byte chunks per lane. ldk in bf16 elements; base points
// at element (row0, k0).
// ---------------------------------------------------------------------------
__device__ __forceinline__ v16bf load_frag_a(const bf16_t* base, int ldk) {
    int lane = threadIdx.x & 31;
    int m = lane & 15, hf = lane >> 4;
    const char* p = (const char*)(base + (size_t)m * ldk) + hf * 16;
    union { int4 v[2]; v16bf f; } u;
    u.v[0] = *(const int4*)(p);
    u.v[1] = *(const int4*)(p + 32);
    return u.f;
}

__device__ __forceinline__ v16bf load_frag_b(const bf16_t* base, int ldk) {
    int lane = threadIdx.x & 31;
    int n = lane & 15, hf = lane >> 4;
    const char* p = (const char*)(base + (size_t)n * ldk) + hf * 16;
    union { int4 v[2]; v16bf f; } u;
    u.v[0] = *(const int4*)(p);
    u.v[1] = *(const int4*)(p + 32);
    return u.f;
}

#define WMMA_BF16(a, b, c) \
    __builtin_amdgcn_wmma_f32_16x16x32_bf16(false, (a), false, (b), (short)0, (c), false, false)

// ---------------------------------------------------------------------------
// adaLN: mod[b][j] = sum_i c[b][i] * W[i][j] + bias[j]   (2 x 256 @ 256 x 8192)
// ---------------------------------------------------------------------------
__global__ __launch_bounds__(256) void adaln_kernel(const float* __restrict__ c,
                                                    const float* __restrict__ W,
                                                    const float* __restrict__ bias,
                                                    float* __restrict__ mod) {
    int j = blockIdx.x * 256 + threadIdx.x;
    if (j >= 4 * Dc) return;
    for (int b = 0; b < Bc; ++b) {
        float s = bias[j];
        for (int i = 0; i < ADALNc; ++i)
            s += c[b * ADALNc + i] * W[(size_t)i * (4 * Dc) + j];
        mod[(size_t)b * (4 * Dc) + j] = s;
    }
}

// ---------------------------------------------------------------------------
// Weight convert + transpose: W[K][N] f32 -> Wt[Npad][Kpad] bf16 (zero padded)
// ---------------------------------------------------------------------------
__global__ __launch_bounds__(256) void transpose_convert_kernel(const float* __restrict__ W,
                                                                bf16_t* __restrict__ Wt,
                                                                int K, int N, int Kpad, int Npad) {
    __shared__ float tile[32][33];
    int bk = blockIdx.x * 32, bn = blockIdx.y * 32;
    int tx = threadIdx.x & 31, ty = threadIdx.x >> 5;   // 32 x 8
    for (int i = 0; i < 32; i += 8) {
        int k = bk + ty + i, n = bn + tx;
        tile[ty + i][tx] = (k < K && n < N) ? W[(size_t)k * N + n] : 0.f;
    }
    __syncthreads();
    for (int i = 0; i < 32; i += 8) {
        int n = bn + ty + i, k = bk + tx;
        if (n < Npad && k < Kpad)
            Wt[(size_t)n * Kpad + k] = (bf16_t)tile[tx][ty + i];
    }
}

// ---------------------------------------------------------------------------
// Row-wise RMSNorm + adaLN scale -> bf16:  out = rms(x)*w*(1 + mod[b][off+d])
// ---------------------------------------------------------------------------
__global__ __launch_bounds__(256) void rmsnorm_scale_kernel(const float* __restrict__ x,
                                                            const float* __restrict__ w,
                                                            const float* __restrict__ mod,
                                                            int mod_off,
                                                            bf16_t* __restrict__ out) {
    int row = blockIdx.x;
    int b = row >> 11;                 // row / S
    const float* xr = x + (size_t)row * Dc;
    __shared__ float red[8];
    float ss = 0.f;
    for (int d = threadIdx.x; d < Dc; d += 256) { float v = xr[d]; ss += v * v; }
    for (int o = 16; o > 0; o >>= 1) ss += __shfl_xor(ss, o, 32);
    if ((threadIdx.x & 31) == 0) red[threadIdx.x >> 5] = ss;
    __syncthreads();
    float tot = 0.f;
    for (int i = 0; i < 8; ++i) tot += red[i];
    float r = rsqrtf(tot / (float)Dc + EPSc);
    const float* md = mod + (size_t)b * (4 * Dc) + mod_off;
    for (int d = threadIdx.x; d < Dc; d += 256)
        out[(size_t)row * Dc + d] = (bf16_t)(xr[d] * r * w[d] * (1.f + md[d]));
}

// ---------------------------------------------------------------------------
// Gated residual: out = x + tanh(mod[b][gate+d]) * (rms(src)*w)
// ---------------------------------------------------------------------------
__global__ __launch_bounds__(256) void residual_rms_kernel(const float* __restrict__ xin,
                                                           const float* __restrict__ src,
                                                           const float* __restrict__ w,
                                                           const float* __restrict__ mod,
                                                           int gate_off,
                                                           float* __restrict__ out) {
    int row = blockIdx.x;
    int b = row >> 11;
    const float* sr = src + (size_t)row * Dc;
    __shared__ float red[8];
    float ss = 0.f;
    for (int d = threadIdx.x; d < Dc; d += 256) { float v = sr[d]; ss += v * v; }
    for (int o = 16; o > 0; o >>= 1) ss += __shfl_xor(ss, o, 32);
    if ((threadIdx.x & 31) == 0) red[threadIdx.x >> 5] = ss;
    __syncthreads();
    float tot = 0.f;
    for (int i = 0; i < 8; ++i) tot += red[i];
    float r = rsqrtf(tot / (float)Dc + EPSc);
    const float* gd = mod + (size_t)b * (4 * Dc) + gate_off;
    for (int d = threadIdx.x; d < Dc; d += 256)
        out[(size_t)row * Dc + d] =
            xin[(size_t)row * Dc + d] + tanhf(gd[d]) * (sr[d] * r * w[d]);
}

// ---------------------------------------------------------------------------
// Main bf16 WMMA GEMM: C[M][Npad] f32 = A[M][K] bf16 * Bt[Npad][K]^T bf16
// Block tile 128x128, 8 waves each computing 32x64. Double-buffered LDS,
// staged with CDNA5 async global->LDS copies so DMA overlaps WMMA issue.
// K multiple of 32, M multiple of 128, Npad multiple of 128.
// ---------------------------------------------------------------------------
__global__ __launch_bounds__(256) void gemm_bf16_kernel(const bf16_t* __restrict__ A,
                                                        const bf16_t* __restrict__ Bt,
                                                        float* __restrict__ C,
                                                        int K, int Npad) {
    __shared__ bf16_t As[2][128 * 40];   // row stride 40 bf16 (80B): conflict-free frags
    __shared__ bf16_t Bs[2][128 * 40];
    int tid = threadIdx.x;
    int lane = tid & 31, wid = tid >> 5;
    int wm = wid >> 1, wn = wid & 1;     // 4 x 2 wave grid
    int hf = lane >> 4, l15 = lane & 15;

    const bf16_t* Ag = A  + (size_t)blockIdx.y * 128 * K;
    const bf16_t* Bg = Bt + (size_t)blockIdx.x * 128 * K;

    // this thread's two 16B chunks of each 128x32 tile (512 chunks total)
    int r0 = tid >> 2,          c0 = tid & 3;
    int r1 = (tid + 256) >> 2,  c1 = (tid + 256) & 3;
    const bf16_t* Ag0 = Ag + (size_t)r0 * K + c0 * 8;
    const bf16_t* Ag1 = Ag + (size_t)r1 * K + c1 * 8;
    const bf16_t* Bg0 = Bg + (size_t)r0 * K + c0 * 8;
    const bf16_t* Bg1 = Bg + (size_t)r1 * K + c1 * 8;
    int la0 = r0 * 40 + c0 * 8;
    int la1 = r1 * 40 + c1 * 8;

    v8f acc[2][4];
    for (int i = 0; i < 2; ++i)
        for (int j = 0; j < 4; ++j) acc[i][j] = v8f_zero();

    // prologue: stage k-tile 0 into buffer 0
    async_copy_b128(As[0] + la0, Ag0);
    async_copy_b128(As[0] + la1, Ag1);
    async_copy_b128(Bs[0] + la0, Bg0);
    async_copy_b128(Bs[0] + la1, Bg1);

    int nk = K >> 5;
    for (int kt = 0; kt < nk; ++kt) {
        int cur = kt & 1;
        wait_async();           // own copies (incl. the ones for `cur`) done
        __syncthreads();        // everyone's copies visible; prev reads retired
        if (kt + 1 < nk) {      // stage next tile while we compute this one
            int k1 = (kt + 1) << 5;
            async_copy_b128(As[cur ^ 1] + la0, Ag0 + k1);
            async_copy_b128(As[cur ^ 1] + la1, Ag1 + k1);
            async_copy_b128(Bs[cur ^ 1] + la0, Bg0 + k1);
            async_copy_b128(Bs[cur ^ 1] + la1, Bg1 + k1);
        }
        v16bf af[2], bfr[4];
        for (int i = 0; i < 2; ++i)
            af[i] = load_frag_a(As[cur] + (wm * 32 + i * 16) * 40, 40);
        for (int j = 0; j < 4; ++j)
            bfr[j] = load_frag_b(Bs[cur] + (wn * 64 + j * 16) * 40, 40);
        for (int i = 0; i < 2; ++i)
            for (int j = 0; j < 4; ++j)
                acc[i][j] = WMMA_BF16(af[i], bfr[j], acc[i][j]);
    }

    // store C: VGPR r -> row (hf? +8 : +0)+r, col = l15
    for (int i = 0; i < 2; ++i)
        for (int j = 0; j < 4; ++j)
            for (int r = 0; r < 8; ++r) {
                int row = blockIdx.y * 128 + wm * 32 + i * 16 + hf * 8 + r;
                int col = blockIdx.x * 128 + wn * 64 + j * 16 + l15;
                C[(size_t)row * Npad + col] = acc[i][j][r];
            }
}

// ---------------------------------------------------------------------------
// Per-head RMSNorm + RoPE for q,k; writes bf16 [B][H][S][HD]
// ---------------------------------------------------------------------------
__global__ __launch_bounds__(128) void qk_rope_kernel(const float* __restrict__ qraw,
                                                      const float* __restrict__ kraw,
                                                      const float* __restrict__ nqw,
                                                      const float* __restrict__ nkw,
                                                      const float* __restrict__ cosb,
                                                      const float* __restrict__ sinb,
                                                      bf16_t* __restrict__ qb,
                                                      bf16_t* __restrict__ kb) {
    int idx = blockIdx.x;
    int bs = idx >> 4;         // b*S + s
    int h  = idx & 15;
    int b  = bs >> 11, s = bs & 2047;
    int t  = threadIdx.x;
    const float* q = qraw + (size_t)bs * Dc + h * HDc;
    const float* k = kraw + (size_t)bs * Dc + h * HDc;
    __shared__ float qn[128], kn[128], redq[4], redk[4];
    float qv = q[t], kv = k[t];
    float sq = qv * qv, sk = kv * kv;
    for (int o = 16; o > 0; o >>= 1) { sq += __shfl_xor(sq, o, 32); sk += __shfl_xor(sk, o, 32); }
    if ((t & 31) == 0) { redq[t >> 5] = sq; redk[t >> 5] = sk; }
    __syncthreads();
    float tq = redq[0] + redq[1] + redq[2] + redq[3];
    float tk = redk[0] + redk[1] + redk[2] + redk[3];
    float rq = rsqrtf(tq / (float)HDc + EPSc);
    float rk = rsqrtf(tk / (float)HDc + EPSc);
    qn[t] = qv * rq * nqw[t];
    kn[t] = kv * rk * nkw[t];
    __syncthreads();
    if (t < 64) {
        float cc = cosb[(size_t)bs * 64 + t];
        float sn = sinb[(size_t)bs * 64 + t];
        size_t o = ((size_t)(b * Hc + h) * Sc + s) * HDc;
        float qr = qn[2 * t], qi = qn[2 * t + 1];
        float kr = kn[2 * t], ki = kn[2 * t + 1];
        qb[o + 2 * t]     = (bf16_t)(qr * cc - qi * sn);
        qb[o + 2 * t + 1] = (bf16_t)(qr * sn + qi * cc);
        kb[o + 2 * t]     = (bf16_t)(kr * cc - ki * sn);
        kb[o + 2 * t + 1] = (bf16_t)(kr * sn + ki * cc);
    }
}

// V f32 [B*S][D] -> Vt bf16 [B][H][HD][S]  (transposed for PV WMMA B operand)
__global__ __launch_bounds__(256) void v_transpose_kernel(const float* __restrict__ vraw,
                                                          bf16_t* __restrict__ vt) {
    size_t i = (size_t)blockIdx.x * 256 + threadIdx.x;
    if (i >= (size_t)Bc * Hc * HDc * Sc) return;
    int s  = (int)(i & 2047);
    int d  = (int)((i >> 11) & 127);
    int bh = (int)(i >> 18);
    int b = bh >> 4, h = bh & 15;
    vt[i] = (bf16_t)vraw[((size_t)(b * Sc + s)) * Dc + h * HDc + d];
}

// ---------------------------------------------------------------------------
// Flash attention (non-causal), bf16 WMMA, f32 accumulate.
// grid = (S/128, B*H); 8 waves; each wave owns 16 query rows.
// Key blocks of 64; K/V staged via async global->LDS copies.
// ---------------------------------------------------------------------------
__global__ __launch_bounds__(256) void flash_attn_kernel(const bf16_t* __restrict__ qg,
                                                         const bf16_t* __restrict__ kg,
                                                         const bf16_t* __restrict__ vtg,
                                                         bf16_t* __restrict__ outg) {
    __shared__ bf16_t Ks[64 * 136];        // 64 keys x 128 hd (stride 136)
    __shared__ bf16_t Vs[128 * 72];        // 128 hd  x 64 keys (stride 72)
    __shared__ bf16_t Ps[8 * 16 * 72];     // per-wave P tile 16 x 64 (stride 72)

    int tid = threadIdx.x, lane = tid & 31, wid = tid >> 5;
    int hf = lane >> 4, l15 = lane & 15;
    int qblk = blockIdx.x;
    int bh = blockIdx.y;
    int b = bh >> 4, h = bh & 15;
    int qrow0 = qblk * 128 + wid * 16;

    const bf16_t* qbase = qg + ((size_t)bh * Sc + qrow0) * HDc;
    v16bf qf[4];
    for (int kk = 0; kk < 4; ++kk)
        qf[kk] = load_frag_a(qbase + kk * 32, HDc);

    v8f ao[8];
    for (int t = 0; t < 8; ++t) ao[t] = v8f_zero();
    float mr[8], lr[8];
    for (int r = 0; r < 8; ++r) { mr[r] = -1e30f; lr[r] = 0.f; }
    const float scale = 0.08838834764831845f;   // 1/sqrt(128)
    bf16_t* Pw = Ps + wid * 16 * 72;

    for (int k0 = 0; k0 < Sc; k0 += 64) {
        __syncthreads();                   // prior iteration done reading Ks/Vs
        {   // K block: 64 rows x 256B; V block: 128 rows x 128B (async DMA)
            const bf16_t* kgb = kg + ((size_t)bh * Sc + k0) * HDc;
            for (int i = 0; i < 4; ++i) {
                int cch = tid + i * 256;
                int row = cch >> 4, col = cch & 15;
                async_copy_b128(Ks + row * 136 + col * 8,
                                kgb + (size_t)row * HDc + col * 8);
            }
            const bf16_t* vgb = vtg + (size_t)bh * HDc * Sc + k0;
            for (int i = 0; i < 4; ++i) {
                int cch = tid + i * 256;
                int row = cch >> 3, col = cch & 7;
                async_copy_b128(Vs + row * 72 + col * 8,
                                vgb + (size_t)row * Sc + col * 8);
            }
            wait_async();
        }
        __syncthreads();

        // scores: 16 q-rows x 64 keys
        v8f sc[4];
        for (int j = 0; j < 4; ++j) {
            v8f a = v8f_zero();
            for (int kk = 0; kk < 4; ++kk) {
                v16bf bfr = load_frag_b(Ks + (j * 16) * 136 + kk * 32, 136);
                a = WMMA_BF16(qf[kk], bfr, a);
            }
            sc[j] = a;
        }

        // online softmax per row (rows live per VGPR slot r, 16-lane groups)
        float corr[8];
        for (int r = 0; r < 8; ++r) {
            float mx = sc[0][r];
            for (int j = 1; j < 4; ++j) mx = fmaxf(mx, sc[j][r]);
            for (int o = 8; o > 0; o >>= 1) mx = fmaxf(mx, __shfl_xor(mx, o, 32));
            mx *= scale;
            float mn = fmaxf(mr[r], mx);
            corr[r] = __expf(mr[r] - mn);
            float rs = 0.f;
            for (int j = 0; j < 4; ++j) {
                float pv = __expf(sc[j][r] * scale - mn);
                rs += pv;
                Pw[(hf * 8 + r) * 72 + j * 16 + l15] = (bf16_t)pv;
            }
            for (int o = 8; o > 0; o >>= 1) rs += __shfl_xor(rs, o, 32);
            lr[r] = lr[r] * corr[r] + rs;
            mr[r] = mn;
        }
        for (int t = 0; t < 8; ++t)
            for (int r = 0; r < 8; ++r) ao[t][r] *= corr[r];

        // O += P @ V  (A = P tile via LDS, B = Vt tile)
        v16bf pf[2];
        for (int kk = 0; kk < 2; ++kk)
            pf[kk] = load_frag_a(Pw + kk * 32, 72);
        for (int t = 0; t < 8; ++t)
            for (int kk = 0; kk < 2; ++kk) {
                v16bf vf = load_frag_b(Vs + (t * 16) * 72 + kk * 32, 72);
                ao[t] = WMMA_BF16(pf[kk], vf, ao[t]);
            }
    }

    // epilogue: divide by l, write merged-head bf16 [B*S][D]
    for (int t = 0; t < 8; ++t)
        for (int r = 0; r < 8; ++r) {
            int s = qrow0 + hf * 8 + r;
            int d = t * 16 + l15;
            outg[((size_t)(b * Sc + s)) * Dc + h * HDc + d] =
                (bf16_t)(ao[t][r] / lr[r]);
        }
}

// g = silu(a1) * a3  -> bf16 (padded tail is silu(0)*0 = 0)
__global__ __launch_bounds__(256) void silu_mul_kernel(const float* __restrict__ a1,
                                                       const float* __restrict__ a3,
                                                       bf16_t* __restrict__ g,
                                                       size_t n) {
    size_t i = (size_t)blockIdx.x * 256 + threadIdx.x;
    if (i >= n) return;
    float x = a1[i];
    float s = x / (1.f + __expf(-x));
    g[i] = (bf16_t)(s * a3[i]);
}

// ---------------------------------------------------------------------------
// Launch
// ---------------------------------------------------------------------------
extern "C" void kernel_launch(void* const* d_in, const int* in_sizes, int n_in,
                              void* d_out, int out_size, void* d_ws, size_t ws_size,
                              hipStream_t stream) {
    (void)in_sizes; (void)n_in; (void)out_size; (void)ws_size;

    const float* x        = (const float*)d_in[0];
    const float* c        = (const float*)d_in[1];
    const float* fcos     = (const float*)d_in[2];
    const float* fsin     = (const float*)d_in[3];
    const float* wq       = (const float*)d_in[4];
    const float* wk       = (const float*)d_in[5];
    const float* wv       = (const float*)d_in[6];
    const float* wo       = (const float*)d_in[7];
    const float* norm_q_w = (const float*)d_in[8];
    const float* norm_k_w = (const float*)d_in[9];
    const float* an1      = (const float*)d_in[10];
    const float* an2      = (const float*)d_in[11];
    const float* fn1      = (const float*)d_in[12];
    const float* fn2      = (const float*)d_in[13];
    const float* w1       = (const float*)d_in[14];
    const float* w2       = (const float*)d_in[15];
    const float* w3       = (const float*)d_in[16];
    const float* adaln_w  = (const float*)d_in[17];
    const float* adaln_b  = (const float*)d_in[18];
    float* out = (float*)d_out;

    // workspace carve-out
    char* p = (char*)d_ws;
    auto alloc = [&](size_t bytes) -> void* {
        void* r = (void*)p;
        p += (bytes + 255) & ~(size_t)255;
        return r;
    };
    float*  mod    = (float*) alloc((size_t)Bc * 4 * Dc * 4);
    bf16_t* hbuf   = (bf16_t*)alloc((size_t)Mrows * Dc * 2);          // h / h2 (reused)
    bf16_t* wqT    = (bf16_t*)alloc((size_t)Dc * Dc * 2);
    bf16_t* wkT    = (bf16_t*)alloc((size_t)Dc * Dc * 2);
    bf16_t* wvT    = (bf16_t*)alloc((size_t)Dc * Dc * 2);
    bf16_t* woT    = (bf16_t*)alloc((size_t)Dc * Dc * 2);
    bf16_t* w1T    = (bf16_t*)alloc((size_t)FFp * Dc * 2);
    bf16_t* w3T    = (bf16_t*)alloc((size_t)FFp * Dc * 2);
    bf16_t* w2T    = (bf16_t*)alloc((size_t)Dc * FFp * 2);
    float*  qraw   = (float*) alloc((size_t)Mrows * Dc * 4);
    float*  kraw   = (float*) alloc((size_t)Mrows * Dc * 4);
    float*  vraw   = (float*) alloc((size_t)Mrows * Dc * 4);
    bf16_t* qb     = (bf16_t*)alloc((size_t)Mrows * Dc * 2);
    bf16_t* kb     = (bf16_t*)alloc((size_t)Mrows * Dc * 2);
    bf16_t* vt     = (bf16_t*)alloc((size_t)Mrows * Dc * 2);
    bf16_t* attnb  = (bf16_t*)alloc((size_t)Mrows * Dc * 2);
    float*  attnp  = (float*) alloc((size_t)Mrows * Dc * 4);
    float*  x1     = (float*) alloc((size_t)Mrows * Dc * 4);
    float*  a1     = (float*) alloc((size_t)Mrows * FFp * 4);
    float*  a3     = (float*) alloc((size_t)Mrows * FFp * 4);
    bf16_t* g      = (bf16_t*)alloc((size_t)Mrows * FFp * 2);
    float*  ff     = (float*) alloc((size_t)Mrows * Dc * 4);

    // 1. adaLN modulation
    adaln_kernel<<<(4 * Dc) / 256, 256, 0, stream>>>(c, adaln_w, adaln_b, mod);

    // 2. weight conversion (f32 -> bf16, transposed to [N][K])
    dim3 t64(64, 64);
    transpose_convert_kernel<<<t64, 256, 0, stream>>>(wq, wqT, Dc, Dc, Dc, Dc);
    transpose_convert_kernel<<<t64, 256, 0, stream>>>(wk, wkT, Dc, Dc, Dc, Dc);
    transpose_convert_kernel<<<t64, 256, 0, stream>>>(wv, wvT, Dc, Dc, Dc, Dc);
    transpose_convert_kernel<<<t64, 256, 0, stream>>>(wo, woT, Dc, Dc, Dc, Dc);
    transpose_convert_kernel<<<dim3(64, 172), 256, 0, stream>>>(w1, w1T, Dc, FFc, Dc, FFp);
    transpose_convert_kernel<<<dim3(64, 172), 256, 0, stream>>>(w3, w3T, Dc, FFc, Dc, FFp);
    transpose_convert_kernel<<<dim3(172, 64), 256, 0, stream>>>(w2, w2T, FFc, Dc, FFp, Dc);

    // 3. h = rms(x) * (1 + scale_msa)
    rmsnorm_scale_kernel<<<Mrows, 256, 0, stream>>>(x, an1, mod, 0, hbuf);

    // 4. QKV projections (bf16 WMMA GEMMs)
    dim3 gproj(Dc / 128, Mrows / 128);
    gemm_bf16_kernel<<<gproj, 256, 0, stream>>>(hbuf, wqT, qraw, Dc, Dc);
    gemm_bf16_kernel<<<gproj, 256, 0, stream>>>(hbuf, wkT, kraw, Dc, Dc);
    gemm_bf16_kernel<<<gproj, 256, 0, stream>>>(hbuf, wvT, vraw, Dc, Dc);

    // 5. per-head RMSNorm + RoPE, V transpose
    qk_rope_kernel<<<Bc * Sc * Hc, 128, 0, stream>>>(qraw, kraw, norm_q_w, norm_k_w,
                                                     fcos, fsin, qb, kb);
    v_transpose_kernel<<<(Bc * Hc * HDc * Sc) / 256, 256, 0, stream>>>(vraw, vt);

    // 6. flash attention
    flash_attn_kernel<<<dim3(Sc / 128, Bc * Hc), 256, 0, stream>>>(qb, kb, vt, attnb);

    // 7. out projection + gated residual 1
    gemm_bf16_kernel<<<gproj, 256, 0, stream>>>(attnb, woT, attnp, Dc, Dc);
    residual_rms_kernel<<<Mrows, 256, 0, stream>>>(x, attnp, an2, mod, Dc, x1);

    // 8. FFN
    rmsnorm_scale_kernel<<<Mrows, 256, 0, stream>>>(x1, fn1, mod, 2 * Dc, hbuf);
    dim3 gff(FFp / 128, Mrows / 128);
    gemm_bf16_kernel<<<gff, 256, 0, stream>>>(hbuf, w1T, a1, Dc, FFp);
    gemm_bf16_kernel<<<gff, 256, 0, stream>>>(hbuf, w3T, a3, Dc, FFp);
    size_t ng = (size_t)Mrows * FFp;
    silu_mul_kernel<<<(unsigned)((ng + 255) / 256), 256, 0, stream>>>(a1, a3, g, ng);
    gemm_bf16_kernel<<<gproj, 256, 0, stream>>>(g, w2T, ff, FFp, Dc);

    // 9. gated residual 2 -> output
    residual_rms_kernel<<<Mrows, 256, 0, stream>>>(x1, ff, fn2, mod, 3 * Dc, out);
}
